// uvtranse_classifier_9766755631661
// MI455X (gfx1250) — matrix operand order repaired and checked by reading, not compiled, not run on hardware
//
#include <hip/hip_runtime.h>
#include <hip/hip_bf16.h>

typedef __attribute__((ext_vector_type(16))) __bf16 v16bf;
typedef __attribute__((ext_vector_type(8)))  __bf16 v8bf;
typedef __attribute__((ext_vector_type(8)))  float  v8f;
typedef __attribute__((ext_vector_type(4)))  float  v4f;

#define N_IM   16
#define N_OBJ  16
#define D_IN   2048
#define N_PAIR 3840      // 16*16*15
#define NUM_REL 3

__device__ __forceinline__ __bf16 f2bf(float f) {
    unsigned u = __builtin_bit_cast(unsigned, f);
    u += 0x7FFFu + ((u >> 16) & 1u);                 // round-to-nearest-even
    unsigned short h = (unsigned short)(u >> 16);
    return __builtin_bit_cast(__bf16, h);
}

// ---------------------------------------------------------------------------
// Weight prep: W (K x N, f32 row-major)  ->  Wt (N x Kpad, bf16 row-major),
// zero-padded for k in [K, Kpad).
// ---------------------------------------------------------------------------
__global__ void transpose_w_bf16(const float* __restrict__ W, __bf16* __restrict__ Wt,
                                 int K, int N, int Kpad) {
    int t = blockIdx.x * blockDim.x + threadIdx.x;
    if (t >= N * Kpad) return;
    int n = t / Kpad;
    int k = t - n * Kpad;
    float v = (k < K) ? W[(size_t)k * N + n] : 0.0f;
    Wt[t] = f2bf(v);
}

// ---------------------------------------------------------------------------
// WMMA GEMM:  C = relu(A @ Bt^T + bias)
//   A : M x K  (f32 if A_F32 else bf16), row-major, row stride K
//   Bt: N x K  bf16 row-major (pre-transposed weights), row stride K
//   C : M x N  (f32 if OUT_F32 else bf16), row-major
// One wave computes a 32 (M) x 64 (N) tile: 2 A-frags x 4 B-frags = 8 wmma
// per k-step of 32.  M % 32 == 0, N % 64 == 0, K % 32 == 0.
// ---------------------------------------------------------------------------
template <bool A_F32, bool OUT_F32>
__global__ __launch_bounds__(256)
void wmma_gemm_relu(const void* __restrict__ Av, const __bf16* __restrict__ Bt,
                    const float* __restrict__ bias, void* __restrict__ Cv,
                    int M, int N, int K) {
    const int lane = threadIdx.x & 31;
    const int wave = blockIdx.x * (blockDim.x >> 5) + (threadIdx.x >> 5);
    const int mtiles  = M >> 5;                 // 32-row tiles
    const int ngroups = N >> 6;                 // 64-col groups
    if (wave >= mtiles * ngroups) return;       // uniform per-wave exit

    const int mt = wave % mtiles;
    const int ng = wave / mtiles;
    const int m0 = mt << 5;
    const int n0 = ng << 6;

    const int l15 = lane & 15;
    const int kbA = (lane >> 4) << 3;   // A k-base: 0 or 8
    const int kbB = (lane >> 4) << 3;   // B k-base: 0 or 8

    const int rowA0 = m0 + l15;         // rows for the two A fragments
    const int rowA1 = m0 + 16 + l15;

    v8f acc[2][4] = {};

    for (int k0 = 0; k0 < K; k0 += 32) {
        // ---- two A fragments: k offsets {0..7, 16..23} from (k0 + kbA) ----
        v16bf a[2];
        #pragma unroll
        for (int i = 0; i < 2; ++i) {
            const int row = i ? rowA1 : rowA0;
            if (A_F32) {
                const float* ap = (const float*)Av + (size_t)row * K + k0 + kbA;
                const v4f* p = (const v4f*)ap;
                v4f f0 = p[0], f1 = p[1], f2 = p[4], f3 = p[5];
                if (k0 + 64 <= K)                       // prefetch next k-block
                    __builtin_prefetch(ap + 32, 0, 1);  // -> global_prefetch_b8
                #pragma unroll
                for (int e = 0; e < 4; ++e) {
                    a[i][e]      = f2bf(f0[e]);
                    a[i][4 + e]  = f2bf(f1[e]);
                    a[i][8 + e]  = f2bf(f2[e]);
                    a[i][12 + e] = f2bf(f3[e]);
                }
            } else {
                const __bf16* ap = (const __bf16*)Av + (size_t)row * K + k0 + kbA;
                const v8bf* p = (const v8bf*)ap;
                v8bf lo = p[0], hi = p[2];     // elements 0..7 and 16..23
                #pragma unroll
                for (int e = 0; e < 8; ++e) { a[i][e] = lo[e]; a[i][8 + e] = hi[e]; }
            }
        }

        // ---- 4 B fragments, each feeding 2 wmma (one per A fragment) ----
        #pragma unroll
        for (int j = 0; j < 4; ++j) {
            const __bf16* bp = Bt + (size_t)(n0 + j * 16 + l15) * K + k0 + kbB;
            const v8bf* q = (const v8bf*)bp;
            v8bf lo = q[0], hi = q[2];         // elements 0..7 and 16..23
            v16bf b;
            #pragma unroll
            for (int e = 0; e < 8; ++e) { b[e] = lo[e]; b[8 + e] = hi[e]; }
            acc[0][j] = __builtin_amdgcn_wmma_f32_16x16x32_bf16(
                            false, a[0], false, b, (short)0, acc[0][j], false, false);
            acc[1][j] = __builtin_amdgcn_wmma_f32_16x16x32_bf16(
                            false, a[1], false, b, (short)0, acc[1][j], false, false);
        }
    }

    // ---- epilogue: bias + relu, store ----
    #pragma unroll
    for (int i = 0; i < 2; ++i) {
        const int rbase = m0 + 16 * i + ((lane >> 4) << 3);
        #pragma unroll
        for (int j = 0; j < 4; ++j) {
            const int col = n0 + j * 16 + l15;
            const float bv = bias ? bias[col] : 0.0f;
            #pragma unroll
            for (int r = 0; r < 8; ++r) {
                float v = acc[i][j][r] + bv;
                v = v > 0.0f ? v : 0.0f;
                const size_t idx = (size_t)(rbase + r) * N + col;
                if (OUT_F32) ((float*)Cv)[idx] = v;
                else         ((__bf16*)Cv)[idx] = f2bf(v);
            }
        }
    }
}

// ---------------------------------------------------------------------------
// appear = fu - fo - fs  ->  Z[:, 0:256] (bf16), Z row stride 288
// ---------------------------------------------------------------------------
__global__ void appear_concat(const float* __restrict__ Fs, const float* __restrict__ Fo,
                              const float* __restrict__ Fu, __bf16* __restrict__ Z) {
    int t = blockIdx.x * blockDim.x + threadIdx.x;
    if (t >= N_PAIR * 256) return;
    int p = t >> 8, c = t & 255;
    Z[(size_t)p * 288 + c] = f2bf(Fu[t] - Fo[t] - Fs[t]);
}

// ---------------------------------------------------------------------------
// 19-dim geometric features -> Z[:, 256:275] (bf16); zero Z[:, 275:288]
// ---------------------------------------------------------------------------
__global__ void loc_encode(const float* __restrict__ boxes, const float* __restrict__ im_info,
                           const int* __restrict__ o1, const int* __restrict__ o2,
                           const int* __restrict__ uu, const int* __restrict__ im,
                           __bf16* __restrict__ Z) {
    int p = blockIdx.x * blockDim.x + threadIdx.x;
    if (p >= N_PAIR) return;
    const float* sb = boxes + (size_t)o1[p] * 5 + 1;
    const float* ob = boxes + (size_t)o2[p] * 5 + 1;
    const float* ub = boxes + (size_t)uu[p] * 5 + 1;
    const float* hw = im_info + (size_t)im[p] * 3;
    float h_i = hw[0], w_i = hw[1];
    float x_s = (sb[0] + sb[2]) * 0.5f, y_s = (sb[1] + sb[3]) * 0.5f;
    float w_s = sb[2] - sb[0],          h_s = sb[3] - sb[1];
    float x_o = (ob[0] + ob[2]) * 0.5f, y_o = (ob[1] + ob[3]) * 0.5f;
    float w_o = ob[2] - ob[0],          h_o = ob[3] - ob[1];
    float a_s = w_s * h_s, a_o = w_o * h_o;
    float a_u = (ub[2] - ub[0]) * (ub[3] - ub[1]);
    float area = w_i * h_i;

    float f[19] = {
        x_s / w_i, y_s / h_i, (x_s + w_s) / w_i, (y_s + h_s) / h_i, a_s / area,
        x_o / w_i, y_o / h_i, (x_o + w_o) / w_i, (y_o + h_o) / h_i, a_o / area,
        (x_s - x_o) / w_o, (y_s - y_o) / h_o, __logf(w_s / w_o), __logf(h_s / h_o),
        (x_o - x_s) / w_s, (y_o - y_s) / h_s, __logf(w_o / w_s), __logf(h_o / h_s),
        a_u / area
    };
    __bf16* z = Z + (size_t)p * 288 + 256;
    #pragma unroll
    for (int j = 0; j < 19; ++j) z[j] = f2bf(f[j]);
    const __bf16 zero = __builtin_bit_cast(__bf16, (unsigned short)0);
    #pragma unroll
    for (int j = 19; j < 32; ++j) z[j] = zero;
}

// ---------------------------------------------------------------------------
// out[p, r] = dot(H2[p, :256], Wc2[:, r]) + bc2[r]
// ---------------------------------------------------------------------------
__global__ void final_proj(const float* __restrict__ H2, const float* __restrict__ Wc2,
                           const float* __restrict__ bc2, float* __restrict__ out) {
    int t = blockIdx.x * blockDim.x + threadIdx.x;
    if (t >= N_PAIR * NUM_REL) return;
    int p = t / NUM_REL, r = t - p * NUM_REL;
    const v4f* h = (const v4f*)(H2 + (size_t)p * 256);
    float s = bc2[r];
    #pragma unroll 4
    for (int k4 = 0; k4 < 64; ++k4) {
        v4f hv = h[k4];
        #pragma unroll
        for (int e = 0; e < 4; ++e)
            s = fmaf(hv[e], Wc2[(k4 * 4 + e) * NUM_REL + r], s);
    }
    out[t] = s;
}

// ---------------------------------------------------------------------------
extern "C" void kernel_launch(void* const* d_in, const int* in_sizes, int n_in,
                              void* d_out, int out_size, void* d_ws, size_t ws_size,
                              hipStream_t stream) {
    const float* xs    = (const float*)d_in[0];
    const float* xo    = (const float*)d_in[1];
    const float* xu    = (const float*)d_in[2];
    const float* boxes = (const float*)d_in[3];
    const float* imnfo = (const float*)d_in[4];
    const int*   o1i   = (const int*)d_in[5];
    const int*   o2i   = (const int*)d_in[6];
    const int*   uii   = (const int*)d_in[7];
    const int*   imi   = (const int*)d_in[8];
    const float* Ws1 = (const float*)d_in[9];  const float* bs1 = (const float*)d_in[10];
    const float* Ws2 = (const float*)d_in[11]; const float* bs2 = (const float*)d_in[12];
    const float* Wo1 = (const float*)d_in[13]; const float* bo1 = (const float*)d_in[14];
    const float* Wo2 = (const float*)d_in[15]; const float* bo2 = (const float*)d_in[16];
    const float* Wu1 = (const float*)d_in[17]; const float* bu1 = (const float*)d_in[18];
    const float* Wu2 = (const float*)d_in[19]; const float* bu2 = (const float*)d_in[20];
    const float* Wc1 = (const float*)d_in[21]; const float* bc1 = (const float*)d_in[22];
    const float* Wc2 = (const float*)d_in[23]; const float* bc2 = (const float*)d_in[24];
    float* out = (float*)d_out;

    // ---- workspace carve-up (all regions 256B aligned) ----
    size_t off = 0;
    auto take = [&](size_t bytes) -> void* {
        void* p = (char*)d_ws + off;
        off += (bytes + 255) & ~(size_t)255;
        return p;
    };
    __bf16* W1t_s = (__bf16*)take((size_t)512 * 2048 * 2);
    __bf16* W1t_o = (__bf16*)take((size_t)512 * 2048 * 2);
    __bf16* W1t_u = (__bf16*)take((size_t)512 * 2048 * 2);
    __bf16* W2t_s = (__bf16*)take((size_t)256 * 512 * 2);
    __bf16* W2t_o = (__bf16*)take((size_t)256 * 512 * 2);
    __bf16* W2t_u = (__bf16*)take((size_t)256 * 512 * 2);
    __bf16* Wc1t  = (__bf16*)take((size_t)256 * 288 * 2);
    __bf16* H1_s  = (__bf16*)take((size_t)N_PAIR * 512 * 2);
    __bf16* H1_o  = (__bf16*)take((size_t)N_PAIR * 512 * 2);
    __bf16* H1_u  = (__bf16*)take((size_t)N_PAIR * 512 * 2);
    float*  F_s   = (float*) take((size_t)N_PAIR * 256 * 4);
    float*  F_o   = (float*) take((size_t)N_PAIR * 256 * 4);
    float*  F_u   = (float*) take((size_t)N_PAIR * 256 * 4);
    __bf16* Z     = (__bf16*)take((size_t)N_PAIR * 288 * 2);
    float*  H2    = (float*) take((size_t)N_PAIR * 256 * 4);

    // ---- 1) weight transpose + bf16 convert ----
    {
        int n1 = 512 * 2048;   // W1t
        transpose_w_bf16<<<(n1 + 255) / 256, 256, 0, stream>>>(Ws1, W1t_s, 2048, 512, 2048);
        transpose_w_bf16<<<(n1 + 255) / 256, 256, 0, stream>>>(Wo1, W1t_o, 2048, 512, 2048);
        transpose_w_bf16<<<(n1 + 255) / 256, 256, 0, stream>>>(Wu1, W1t_u, 2048, 512, 2048);
        int n2 = 256 * 512;    // W2t
        transpose_w_bf16<<<(n2 + 255) / 256, 256, 0, stream>>>(Ws2, W2t_s, 512, 256, 512);
        transpose_w_bf16<<<(n2 + 255) / 256, 256, 0, stream>>>(Wo2, W2t_o, 512, 256, 512);
        transpose_w_bf16<<<(n2 + 255) / 256, 256, 0, stream>>>(Wu2, W2t_u, 512, 256, 512);
        int n3 = 256 * 288;    // Wc1t (K padded 275 -> 288)
        transpose_w_bf16<<<(n3 + 255) / 256, 256, 0, stream>>>(Wc1, Wc1t, 275, 256, 288);
    }

    auto gemm_blocks = [](int M, int N) {
        int waves = (M / 32) * (N / 64);
        return (waves + 7) / 8;               // 8 waves / 256-thread block
    };

    // ---- 2) layer 1: (3840x2048) @ (2048x512), fp32 A -> bf16 H1 ----
    {
        int nb = gemm_blocks(N_PAIR, 512);
        wmma_gemm_relu<true, false><<<nb, 256, 0, stream>>>(xs, W1t_s, bs1, H1_s, N_PAIR, 512, 2048);
        wmma_gemm_relu<true, false><<<nb, 256, 0, stream>>>(xo, W1t_o, bo1, H1_o, N_PAIR, 512, 2048);
        wmma_gemm_relu<true, false><<<nb, 256, 0, stream>>>(xu, W1t_u, bu1, H1_u, N_PAIR, 512, 2048);
    }

    // ---- 3) layer 2: (3840x512) @ (512x256), bf16 A -> f32 F ----
    {
        int nb = gemm_blocks(N_PAIR, 256);
        wmma_gemm_relu<false, true><<<nb, 256, 0, stream>>>(H1_s, W2t_s, bs2, F_s, N_PAIR, 256, 512);
        wmma_gemm_relu<false, true><<<nb, 256, 0, stream>>>(H1_o, W2t_o, bo2, F_o, N_PAIR, 256, 512);
        wmma_gemm_relu<false, true><<<nb, 256, 0, stream>>>(H1_u, W2t_u, bu2, F_u, N_PAIR, 256, 512);
    }

    // ---- 4) appear + loc -> Z (3840 x 288 bf16, zero-padded) ----
    appear_concat<<<(N_PAIR * 256 + 255) / 256, 256, 0, stream>>>(F_s, F_o, F_u, Z);
    loc_encode<<<(N_PAIR + 255) / 256, 256, 0, stream>>>(boxes, imnfo, o1i, o2i, uii, imi, Z);

    // ---- 5) classifier hidden: (3840x288) @ (288x256) -> relu f32 H2 ----
    wmma_gemm_relu<false, true><<<gemm_blocks(N_PAIR, 256), 256, 0, stream>>>(
        Z, Wc1t, bc1, H2, N_PAIR, 256, 288);

    // ---- 6) final 256 -> 3 projection ----
    final_proj<<<(N_PAIR * NUM_REL + 255) / 256, 256, 0, stream>>>(H2, Wc2, bc2, out);
}
